// dnc_loss_16664473108582
// MI455X (gfx1250) — matrix operand order repaired and checked by reading, not compiled, not run on hardware
//
#include <hip/hip_runtime.h>

typedef __attribute__((ext_vector_type(16))) _Float16 v16h;
typedef __attribute__((ext_vector_type(8)))  float    v8f;

#define BB 8
#define NB 128
#define HH 192
#define WW 256
#define NT 8                      // tiles of 16 boxes
#define HSPLIT 8
#define ROWS_PER_WG (HH / HSPLIT) // 24
#define WAVES 4
#define ROWS_PER_WAVE (ROWS_PER_WG / WAVES) // 6
#define DOWNF 8.0f

__device__ __forceinline__ float hatP(float u) {
    u = fminf(fmaxf(u, -1.0f), 1.0f);
    return (u <= 0.0f) ? 0.5f * (u + 1.0f) * (u + 1.0f)
                       : 0.5f + u - 0.5f * u * u;
}

__global__ __launch_bounds__(128)
void dnc_count_kernel(const float* __restrict__ den,    // [B,1,H,W]
                      const float* __restrict__ hboxes, // [B,N,5]
                      const float* __restrict__ post,   // [B,N,H,W]
                      float* __restrict__ partial)      // [HSPLIT,B,N]
{
    __shared__ float sx1[16], sy1[16], sx2[16], sy2[16];
    __shared__ __align__(16) float wxs[16][WW];      // 16 KB
    __shared__ __align__(16) float wys[16][HH];      // 12 KB
    __shared__ __align__(16) float denrow[WAVES][WW];// 4 KB
    __shared__ float diag[WAVES][16];

    const int tid  = threadIdx.x;
    const int lane = tid & 31;
    const int wv   = tid >> 5;

    int blk = blockIdx.x;
    const int hs = blk % HSPLIT; blk /= HSPLIT;
    const int nt = blk % NT;
    const int b  = blk / NT;

    // ---- box coords (downscaled) into LDS ----
    if (tid < 16) {
        const float* hb = hboxes + (size_t)(b * NB + nt * 16 + tid) * 5;
        sx1[tid] = hb[0] * (1.0f / DOWNF);
        sy1[tid] = hb[1] * (1.0f / DOWNF);
        sx2[tid] = hb[2] * (1.0f / DOWNF);
        sy2[tid] = hb[3] * (1.0f / DOWNF);
    }
    __syncthreads();

    // ---- separable hat-integral weights into LDS ----
    for (int idx = tid; idx < 16 * WW; idx += 128) {
        int n = idx >> 8, w = idx & (WW - 1);
        wxs[n][w] = hatP(sx2[n] - (float)w) - hatP(sx1[n] - (float)w);
    }
    for (int idx = tid; idx < 16 * HH; idx += 128) {
        int n = idx / HH, h = idx % HH;
        wys[n][h] = hatP(sy2[n] - (float)h) - hatP(sy1[n] - (float)h);
    }
    __syncthreads();

    // ---- preload B fragments (wx), h-invariant, kept in 64 VGPRs ----
    // B 32x16 f16 layout: lane l holds column n = l&15; K(j) = (l<16?0:16)+j
    const int nloc  = lane & 15;
    const int koffB = (lane < 16) ? 0 : 16;
    v16h bfrag[8];
#pragma unroll
    for (int c = 0; c < 8; ++c) {
#pragma unroll
        for (int j = 0; j < 16; ++j)
            bfrag[c][j] = (_Float16)wxs[nloc][c * 32 + koffB + j];
    }

    // A 16x32 f16 layout: lane l holds row m = l&15;
    // K(j) = (l<16?0:16)+j
    const int offA = (lane < 16) ? 0 : 8;
    const size_t postRowBase = (size_t)(b * NB + nt * 16 + nloc) * HH * WW;
    const size_t denBase     = (size_t)b * HH * WW;

    // Two independent accumulator chains (even/odd chunks) so the WMMA
    // reading one A buffer never hazards against the VALU building the
    // other -> removes the 4 co-exec v_nops per WMMA (ISA 7.12.1).
    v8f acc0 = {0.f, 0.f, 0.f, 0.f, 0.f, 0.f, 0.f, 0.f};
    v8f acc1 = {0.f, 0.f, 0.f, 0.f, 0.f, 0.f, 0.f, 0.f};

    for (int r = 0; r < ROWS_PER_WAVE; ++r) {
        const int h = hs * ROWS_PER_WG + wv * ROWS_PER_WAVE + r;

        // stage den row into this wave's LDS strip (coalesced b128)
        {
            const float4* dg = (const float4*)(den + denBase + (size_t)h * WW);
            float4* dl = (float4*)&denrow[wv][0];
            dl[lane * 2 + 0] = dg[lane * 2 + 0];
            dl[lane * 2 + 1] = dg[lane * 2 + 1];
        }

        const float  wyn = wys[nloc][h];
        const float* pr  = post + postRowBase + (size_t)h * WW;
        if (r + 1 < ROWS_PER_WAVE)
            __builtin_prefetch(pr + WW + offA, 0, 0);   // global_prefetch next row

#pragma unroll
        for (int c = 0; c < 8; ++c) {
            float pv[16], dv[16];
            *(float4*)&pv[0]  = *(const float4*)(pr + c * 32 + koffB);
            *(float4*)&pv[4]  = *(const float4*)(pr + c * 32 + koffB + 4);
            *(float4*)&pv[8]  = *(const float4*)(pr + c * 32 + koffB + 8);
            *(float4*)&pv[12] = *(const float4*)(pr + c * 32 + koffB + 12);
            *(float4*)&dv[0]  = *(const float4*)(&denrow[wv][c * 32 + koffB]);
            *(float4*)&dv[4]  = *(const float4*)(&denrow[wv][c * 32 + koffB + 4]);
            *(float4*)&dv[8]  = *(const float4*)(&denrow[wv][c * 32 + koffB + 8]);
            *(float4*)&dv[12] = *(const float4*)(&denrow[wv][c * 32 + koffB + 12]);

            v16h a;
#pragma unroll
            for (int j = 0; j < 16; ++j)
                a[j] = (_Float16)(pv[j] * dv[j] * wyn);

            if (c & 1)
                acc1 = __builtin_amdgcn_wmma_f32_16x16x32_f16(
                    false, a, false, bfrag[c], (short)0, acc1, false, false);
            else
                acc0 = __builtin_amdgcn_wmma_f32_16x16x32_f16(
                    false, a, false, bfrag[c], (short)0, acc0, false, false);
        }
    }

    v8f acc = acc0 + acc1;

    // ---- extract diagonal of the 16x16 f32 accumulator ----
    // D[n][n]: n<8 -> lane n, vgpr n ; n>=8 -> lane n+16, vgpr n-8
    int nsel = -1, comp = 0;
    if (lane < 8)       { nsel = lane;      comp = lane; }
    else if (lane >= 24){ nsel = lane - 16; comp = lane - 24; }
    if (nsel >= 0) {
        float dval = 0.0f;
        switch (comp) {
            case 0: dval = acc[0]; break; case 1: dval = acc[1]; break;
            case 2: dval = acc[2]; break; case 3: dval = acc[3]; break;
            case 4: dval = acc[4]; break; case 5: dval = acc[5]; break;
            case 6: dval = acc[6]; break; case 7: dval = acc[7]; break;
        }
        diag[wv][nsel] = dval;
    }
    __syncthreads();

    if (tid < 16) {
        float s = diag[0][tid] + diag[1][tid] + diag[2][tid] + diag[3][tid];
        partial[((size_t)hs * BB + b) * NB + nt * 16 + tid] = s;
    }
}

__global__ __launch_bounds__(32)
void dnc_finalize_kernel(const float* __restrict__ hboxes,
                         const float* __restrict__ partial,
                         float* __restrict__ out)
{
    __shared__ float per_img[BB];
    const int tid = threadIdx.x;
    if (tid < BB) {
        const int b = tid;
        float errsum = 0.0f, num = 0.0f;
        for (int n = 0; n < NB; ++n) {
            float c = 0.0f;
            for (int hsp = 0; hsp < HSPLIT; ++hsp)
                c += partial[((size_t)hsp * BB + b) * NB + n];
            float lab = hboxes[(size_t)(b * NB + n) * 5 + 4];
            float v = (lab > 0.0f) ? 1.0f : 0.0f;
            errsum += fabsf(c - 1.0f) * v;
            num += v;
        }
        per_img[b] = (num > 0.0f) ? (errsum / num) : 0.0f;
    }
    __syncthreads();
    if (tid == 0) {
        float s = 0.0f;
        for (int b = 0; b < BB; ++b) s += per_img[b];
        out[0] = s;
    }
}

extern "C" void kernel_launch(void* const* d_in, const int* in_sizes, int n_in,
                              void* d_out, int out_size, void* d_ws, size_t ws_size,
                              hipStream_t stream) {
    // setup_inputs order: cls(0) reg(1) off(2) den(3) fboxes(4) hboxes(5) ctr(6) post(7)
    const float* den    = (const float*)d_in[3];
    const float* hboxes = (const float*)d_in[5];
    const float* post   = (const float*)d_in[7];
    float* partial = (float*)d_ws;   // HSPLIT*B*N floats = 32 KB

    dnc_count_kernel<<<dim3(BB * NT * HSPLIT), 128, 0, stream>>>(
        den, hboxes, post, partial);
    dnc_finalize_kernel<<<dim3(1), 32, 0, stream>>>(
        hboxes, partial, (float*)d_out);
}
